// LlamaAttention_61186104099186
// MI455X (gfx1250) — matrix operand order repaired and checked by reading, not compiled, not run on hardware
//
#include <hip/hip_runtime.h>
#include <hip/hip_bf16.h>

// ---------------------------------------------------------------------------
// LlamaAttention for MI455X (gfx1250): f16 WMMA (fp32 accumulate) everywhere.
// Memory-bound workload (~230 MB @ 23.3 TB/s); f16 WMMA keeps compute far
// under the memory roofline while fp32-accumulating for accuracy.
// ---------------------------------------------------------------------------

typedef __attribute__((ext_vector_type(16))) _Float16 v16h;
typedef __attribute__((ext_vector_type(8)))  _Float16 v8h;
typedef __attribute__((ext_vector_type(4)))  _Float16 v4h;
typedef __attribute__((ext_vector_type(8)))  float    v8f;
typedef __attribute__((ext_vector_type(4)))  float    v4f;

#define S_LEN 1024
#define HID   4096
#define NH    32
#define NKVH  8
#define HD    128
#define PAST  1024
#define KVLEN (PAST + S_LEN)   // 2048

__device__ __forceinline__ v8f wmma32(v16h a, v16h b, v8f c) {
  // D = A(16x32 f16) * B(32x16 f16) + C(16x16 f32)
  return __builtin_amdgcn_wmma_f32_16x16x32_f16(false, a, false, b,
                                                (short)0, c, false, false);
}
__device__ __forceinline__ v16h join8(v8h lo, v8h hi) {
  v16h r;
#pragma unroll
  for (int i = 0; i < 8; ++i) { r[i] = lo[i]; r[i + 8] = hi[i]; }
  return r;
}
__device__ __forceinline__ v8f zero8() {
  v8f z;
#pragma unroll
  for (int i = 0; i < 8; ++i) z[i] = 0.f;
  return z;
}

// ---------------------------------------------------------------------------
// Generic GEMM: C[M,N] (f32) = A[M,K] (f32 or f16) * B[K,N] (f32)
// Block = 256 threads (8 waves), tile 128(M) x 64(N), K-step 32.
// Each wave computes a 32x32 output patch = 2x2 WMMA tiles.
// A staged row-major in LDS (f16), B staged TRANSPOSED [n][k] so that B
// fragments (N striped across lanes, K contiguous per lane) are 2x b128 reads.
// ---------------------------------------------------------------------------
template <bool A_HALF>
__global__ __launch_bounds__(256) void gemm_wmma_kernel(
    const void* __restrict__ Av, const float* __restrict__ B,
    float* __restrict__ C, int M, int N, int K) {
  __shared__ alignas(16) _Float16 sA[128 * 40];  // 128 rows, K stride 40 (pad)
  __shared__ alignas(16) _Float16 sB[64 * 40];   // [n][k], stride 40

  const int t = threadIdx.x, lane = t & 31, wave = t >> 5;
  const int hs = lane >> 4, l16 = lane & 15;
  const int wm = (wave >> 1) * 32, wn = (wave & 1) * 32;
  const int m0 = blockIdx.y * 128, n0 = blockIdx.x * 64;

  v8f acc[2][2];
#pragma unroll
  for (int mi = 0; mi < 2; ++mi)
#pragma unroll
    for (int ni = 0; ni < 2; ++ni) acc[mi][ni] = zero8();

  for (int k0 = 0; k0 < K; k0 += 32) {
    // ---- stage A tile (128 x 32) -> LDS f16
    if (A_HALF) {
      const _Float16* A = (const _Float16*)Av;
#pragma unroll
      for (int j = 0; j < 2; ++j) {
        int f = t + j * 256;                 // 512 chunks of 8 halves
        int row = f >> 2, c8 = (f & 3) * 8;
        *(v8h*)(sA + row * 40 + c8) =
            *(const v8h*)(A + (size_t)(m0 + row) * K + k0 + c8);
      }
    } else {
      const float* A = (const float*)Av;
#pragma unroll
      for (int j = 0; j < 4; ++j) {
        int f = t + j * 256;                 // 1024 float4 chunks
        int row = f >> 3, c4 = (f & 7) * 4;
        v4f v = *(const v4f*)(A + (size_t)(m0 + row) * K + k0 + c4);
        v4h h;
        h[0] = (_Float16)v[0]; h[1] = (_Float16)v[1];
        h[2] = (_Float16)v[2]; h[3] = (_Float16)v[3];
        *(v4h*)(sA + row * 40 + c4) = h;
      }
    }
    // ---- stage B tile (32 x 64) -> LDS transposed [n][k]
#pragma unroll
    for (int j = 0; j < 2; ++j) {
      int f = t + j * 256;                   // 512 float4 chunks
      int row = f >> 4, c4 = (f & 15) * 4;
      v4f v = *(const v4f*)(B + (size_t)(k0 + row) * N + n0 + c4);
#pragma unroll
      for (int i = 0; i < 4; ++i) sB[(c4 + i) * 40 + row] = (_Float16)v[i];
    }
    // ---- prefetch next tiles into L2 (global_prefetch_b8)
    if (k0 + 32 < K) {
      if (t < 128) {
        const char* Ab = (const char*)Av;
        const size_t esz = A_HALF ? 2 : 4;
        __builtin_prefetch(Ab + ((size_t)(m0 + t) * K + k0 + 32) * esz, 0, 0);
      } else if (t < 192) {
        int r = t - 128;
        __builtin_prefetch(B + (size_t)(k0 + 32 + (r >> 1)) * N + n0 + (r & 1) * 32,
                           0, 0);
      }
    }
    __syncthreads();

    // ---- fragments (layouts per CDNA5 ISA 7.12.2)
    v16h af[2], bf[2];
#pragma unroll
    for (int mi = 0; mi < 2; ++mi) {
      const _Float16* p = sA + (wm + mi * 16 + l16) * 40;
      af[mi] = join8(*(const v8h*)(p + hs * 8), *(const v8h*)(p + 16 + hs * 8));
    }
#pragma unroll
    for (int ni = 0; ni < 2; ++ni) {
      const _Float16* p = sB + (wn + ni * 16 + l16) * 40;
      bf[ni] = join8(*(const v8h*)(p + hs * 8), *(const v8h*)(p + 16 + hs * 8));
    }
#pragma unroll
    for (int mi = 0; mi < 2; ++mi)
#pragma unroll
      for (int ni = 0; ni < 2; ++ni)
        acc[mi][ni] = wmma32(af[mi], bf[ni], acc[mi][ni]);
    __syncthreads();
  }

  // ---- store C (C layout: VGPR r -> M = r + 8*hs, N = l16)
#pragma unroll
  for (int mi = 0; mi < 2; ++mi)
#pragma unroll
    for (int ni = 0; ni < 2; ++ni) {
      int mbase = m0 + wm + mi * 16 + 8 * hs;
      int n = n0 + wn + ni * 16 + l16;
#pragma unroll
      for (int r = 0; r < 8; ++r)
        C[(size_t)(mbase + r) * N + n] = acc[mi][ni][r];
    }
}

// ---------------------------------------------------------------------------
// RoPE Q: Qraw[t, h*128+d] -> Qf16[h][t][d], scale 1/sqrt(128) folded in.
// ---------------------------------------------------------------------------
__global__ void rope_q_kernel(const float* __restrict__ Qraw,
                              _Float16* __restrict__ Qf16) {
  int idx = blockIdx.x * 256 + threadIdx.x;      // 32*1024*64
  int d = idx & 63, tt = (idx >> 6) & 1023, h = idx >> 16;
  float q1 = Qraw[(size_t)tt * HID + h * HD + d];
  float q2 = Qraw[(size_t)tt * HID + h * HD + d + 64];
  float pos = (float)(PAST + tt);
  float inv = expf((float)d * -0.14391156831212787f);   // 10000^(-d/64)
  float fr = pos * inv;
  float c = cosf(fr), s = sinf(fr);
  const float scale = 0.08838834764831845f;              // 1/sqrt(128)
  size_t base = ((size_t)h * S_LEN + tt) * HD;
  Qf16[base + d]      = (_Float16)((q1 * c - q2 * s) * scale);
  Qf16[base + d + 64] = (_Float16)((q2 * c + q1 * s) * scale);
}

// RoPE K + GQA repeat into 32-head cache rows [PAST, PAST+S)
__global__ void rope_k_kernel(const float* __restrict__ Kraw,
                              _Float16* __restrict__ Kc) {
  int idx = blockIdx.x * 256 + threadIdx.x;      // 8*1024*64
  int d = idx & 63, tt = (idx >> 6) & 1023, kh = idx >> 16;
  float k1 = Kraw[(size_t)tt * (NKVH * HD) + kh * HD + d];
  float k2 = Kraw[(size_t)tt * (NKVH * HD) + kh * HD + d + 64];
  float pos = (float)(PAST + tt);
  float inv = expf((float)d * -0.14391156831212787f);
  float fr = pos * inv;
  float c = cosf(fr), s = sinf(fr);
  _Float16 o1 = (_Float16)(k1 * c - k2 * s);
  _Float16 o2 = (_Float16)(k2 * c + k1 * s);
#pragma unroll
  for (int r = 0; r < 4; ++r) {
    int h = kh * 4 + r;
    size_t base = ((size_t)h * KVLEN + PAST + tt) * HD;
    Kc[base + d] = o1;
    Kc[base + d + 64] = o2;
  }
}

// past_key [32][1024][128] f32 -> Kc rows [0, PAST)
__global__ void past_k_kernel(const float* __restrict__ pk,
                              _Float16* __restrict__ Kc) {
  int idx = blockIdx.x * 256 + threadIdx.x;      // 32*1024*128
  int d = idx & 127, tt = (idx >> 7) & 1023, h = idx >> 17;
  Kc[((size_t)h * KVLEN + tt) * HD + d] = (_Float16)pk[idx];
}

// past_value -> Vt[h][d][t] (d-major transpose), rows [0, PAST)
__global__ void past_v_kernel(const float* __restrict__ pv,
                              _Float16* __restrict__ Vt) {
  int idx = blockIdx.x * 256 + threadIdx.x;      // 32*1024*128
  int d = idx & 127, tt = (idx >> 7) & 1023, h = idx >> 17;
  Vt[((size_t)h * HD + d) * KVLEN + tt] = (_Float16)pv[idx];
}

// new V + GQA repeat -> Vt[h][d][PAST+t]
__global__ void new_v_kernel(const float* __restrict__ Vraw,
                             _Float16* __restrict__ Vt) {
  int idx = blockIdx.x * 256 + threadIdx.x;      // 8*1024*128
  int d = idx & 127, tt = (idx >> 7) & 1023, kh = idx >> 17;
  _Float16 v = (_Float16)Vraw[(size_t)tt * (NKVH * HD) + kh * HD + d];
#pragma unroll
  for (int r = 0; r < 4; ++r) {
    int h = kh * 4 + r;
    Vt[((size_t)h * HD + d) * KVLEN + PAST + tt] = v;
  }
}

// ---------------------------------------------------------------------------
// Flash attention: grid (16 q-blocks, 32 heads), 128 threads = 4 waves.
// Each wave owns a 16-row q strip; kv processed in tiles of 32.
// Analytic causal mask (== reference's additive -1e9 mask).
// ---------------------------------------------------------------------------
__global__ __launch_bounds__(128) void flash_attn_kernel(
    const _Float16* __restrict__ Qf16, const _Float16* __restrict__ Kc,
    const _Float16* __restrict__ Vt, _Float16* __restrict__ attn) {
  const int h = blockIdx.y;
  const int qb = blockIdx.x;
  __shared__ alignas(16) _Float16 sK[32 * 136];       // [kv row][d], pad
  __shared__ alignas(16) _Float16 sV[128 * 40];       // [d][kv], pad
  __shared__ alignas(16) _Float16 sP[4 * 16 * 32];    // per-wave P tile

  const int t = threadIdx.x, lane = t & 31, wave = t >> 5;
  const int hs = lane >> 4, l16 = lane & 15;
  const int qrow0 = qb * 64 + wave * 16;
  const int qpos_base = PAST + qrow0;

  // Q fragments held in registers: 16 rows x 128 depth = 4 x v16h
  v16h qf[4];
#pragma unroll
  for (int ds = 0; ds < 4; ++ds) {
    const _Float16* qp = Qf16 + ((size_t)h * S_LEN + qrow0 + l16) * HD + ds * 32;
    qf[ds] = join8(*(const v8h*)(qp + hs * 8), *(const v8h*)(qp + 16 + hs * 8));
  }

  float mrow[8], lrow[8];
  v8f Oacc[8];
#pragma unroll
  for (int r = 0; r < 8; ++r) { mrow[r] = -1e30f; lrow[r] = 0.f; }
#pragma unroll
  for (int dt = 0; dt < 8; ++dt) Oacc[dt] = zero8();

  const int kv_tiles = (PAST + qb * 64 + 64 + 31) / 32;   // cover block's causal span
  for (int kt = 0; kt < kv_tiles; ++kt) {
    const int kv0 = kt * 32;
    __syncthreads();
    // ---- stage K tile (32 x 128) and V tile (128 x 32, d-major) into LDS
#pragma unroll
    for (int j = 0; j < 4; ++j) {
      int f = t + j * 128;
      int row = f >> 4, c8 = (f & 15) * 8;
      *(v8h*)(sK + row * 136 + c8) =
          *(const v8h*)(Kc + ((size_t)h * KVLEN + kv0 + row) * HD + c8);
    }
#pragma unroll
    for (int j = 0; j < 4; ++j) {
      int f = t + j * 128;
      int row = f >> 2, c8 = (f & 3) * 8;
      *(v8h*)(sV + row * 40 + c8) =
          *(const v8h*)(Vt + ((size_t)h * HD + row) * KVLEN + kv0 + c8);
    }
    __syncthreads();

    // ---- scores: S(16x32) = Q(16x128) . K^T, 8 wmma
    v8f sc[2];
    sc[0] = zero8(); sc[1] = zero8();
#pragma unroll
    for (int ds = 0; ds < 4; ++ds) {
#pragma unroll
      for (int nt = 0; nt < 2; ++nt) {
        const _Float16* p = sK + (nt * 16 + l16) * 136 + ds * 32 + hs * 8;
        sc[nt] = wmma32(qf[ds], join8(*(const v8h*)p, *(const v8h*)(p + 16)), sc[nt]);
      }
    }

    // ---- causal mask + online softmax (row = r + 8*hs lives in 16 lanes)
#pragma unroll
    for (int r = 0; r < 8; ++r) {
      int qpos = qpos_base + 8 * hs + r;
#pragma unroll
      for (int nt = 0; nt < 2; ++nt) {
        int kpos = kv0 + nt * 16 + l16;
        if (kpos > qpos) sc[nt][r] += -1e9f;
      }
      float mx = fmaxf(sc[0][r], sc[1][r]);
#pragma unroll
      for (int off = 1; off < 16; off <<= 1)
        mx = fmaxf(mx, __shfl_xor(mx, off));
      float mnew = fmaxf(mrow[r], mx);
      float alpha = expf(mrow[r] - mnew);
      float psum = 0.f;
#pragma unroll
      for (int nt = 0; nt < 2; ++nt) {
        float p = expf(sc[nt][r] - mnew);
        sc[nt][r] = p;
        psum += p;
      }
#pragma unroll
      for (int off = 1; off < 16; off <<= 1)
        psum += __shfl_xor(psum, off);
      lrow[r] = lrow[r] * alpha + psum;
      mrow[r] = mnew;
#pragma unroll
      for (int dt = 0; dt < 8; ++dt) Oacc[dt][r] *= alpha;
    }

    // ---- P (C-layout) -> LDS row-major, reread as A fragment
    _Float16* pw = sP + wave * 512;
#pragma unroll
    for (int nt = 0; nt < 2; ++nt)
#pragma unroll
      for (int r = 0; r < 8; ++r)
        pw[(r + 8 * hs) * 32 + nt * 16 + l16] = (_Float16)sc[nt][r];
    __syncthreads();

    v16h pf = join8(*(const v8h*)(pw + l16 * 32 + hs * 8),
                    *(const v8h*)(pw + l16 * 32 + 16 + hs * 8));
    // ---- O += P(16x32) . V(32x128), 8 wmma
#pragma unroll
    for (int dt = 0; dt < 8; ++dt) {
      const _Float16* p = sV + (dt * 16 + l16) * 40 + hs * 8;
      Oacc[dt] = wmma32(pf, join8(*(const v8h*)p, *(const v8h*)(p + 16)), Oacc[dt]);
    }
  }

  // ---- normalize and store attn[t][h*128+d] (f16, feeds final GEMM)
#pragma unroll
  for (int r = 0; r < 8; ++r) {
    float invl = 1.0f / lrow[r];
    int m = qrow0 + 8 * hs + r;
#pragma unroll
    for (int dt = 0; dt < 8; ++dt) {
      int n = h * HD + dt * 16 + l16;
      attn[(size_t)m * HID + n] = (_Float16)(Oacc[dt][r] * invl);
    }
  }
}

// ---------------------------------------------------------------------------
extern "C" void kernel_launch(void* const* d_in, const int* in_sizes, int n_in,
                              void* d_out, int out_size, void* d_ws, size_t ws_size,
                              hipStream_t stream) {
  (void)in_sizes; (void)n_in; (void)out_size; (void)ws_size;
  const float* X  = (const float*)d_in[0];   // [1,1024,4096]
  const float* pK = (const float*)d_in[1];   // [1,32,1024,128]
  const float* pV = (const float*)d_in[2];   // [1,32,1024,128]
  // d_in[3]: attention_mask — deterministic causal, applied analytically
  const float* Wq = (const float*)d_in[4];   // [4096,4096]
  const float* Wk = (const float*)d_in[5];   // [4096,1024]
  const float* Wv = (const float*)d_in[6];   // [4096,1024]
  const float* Wo = (const float*)d_in[7];   // [4096,4096]

  char* ws = (char*)d_ws;
  const size_t QRAW = (size_t)S_LEN * HID * 4;             // 16.78 MB
  const size_t KRAW = (size_t)S_LEN * NKVH * HD * 4;       //  4.19 MB
  const size_t VRAW = KRAW;
  const size_t QF16 = (size_t)NH * S_LEN * HD * 2;         //  8.39 MB
  const size_t KCB  = (size_t)NH * KVLEN * HD * 2;         // 16.78 MB
  const size_t VTB  = KCB;
  float*    Qraw = (float*)ws;
  float*    Kraw = (float*)(ws + QRAW);
  float*    Vraw = (float*)(ws + QRAW + KRAW);
  _Float16* Qf16 = (_Float16*)(ws + QRAW + KRAW + VRAW);
  _Float16* Kc   = (_Float16*)(ws + QRAW + KRAW + VRAW + QF16);
  _Float16* Vt   = (_Float16*)(ws + QRAW + KRAW + VRAW + QF16 + KCB);
  _Float16* attn = (_Float16*)(ws + QRAW + KRAW + VRAW + QF16 + KCB + VTB);

  // 1) projections
  gemm_wmma_kernel<false><<<dim3(HID / 64, S_LEN / 128), 256, 0, stream>>>(
      X, Wq, Qraw, S_LEN, HID, HID);
  gemm_wmma_kernel<false><<<dim3((NKVH * HD) / 64, S_LEN / 128), 256, 0, stream>>>(
      X, Wk, Kraw, S_LEN, NKVH * HD, HID);
  gemm_wmma_kernel<false><<<dim3((NKVH * HD) / 64, S_LEN / 128), 256, 0, stream>>>(
      X, Wv, Vraw, S_LEN, NKVH * HD, HID);

  // 2) RoPE + KV-cache build
  rope_q_kernel<<<(NH * S_LEN * 64) / 256, 256, 0, stream>>>(Qraw, Qf16);
  rope_k_kernel<<<(NKVH * S_LEN * 64) / 256, 256, 0, stream>>>(Kraw, Kc);
  past_k_kernel<<<(NH * PAST * HD) / 256, 256, 0, stream>>>(pK, Kc);
  past_v_kernel<<<(NH * PAST * HD) / 256, 256, 0, stream>>>(pV, Vt);
  new_v_kernel<<<(NKVH * S_LEN * HD) / 256, 256, 0, stream>>>(Vraw, Vt);

  // 3) flash attention
  flash_attn_kernel<<<dim3(S_LEN / 64, NH), 128, 0, stream>>>(Qf16, Kc, Vt, attn);

  // 4) output projection
  gemm_wmma_kernel<true><<<dim3(HID / 64, S_LEN / 128), 256, 0, stream>>>(
      attn, Wo, (float*)d_out, S_LEN, HID, HID);
}
